// Downsample_PASA_group_softmax_2791728742720
// MI455X (gfx1250) — matrix operand order repaired and checked by reading, not compiled, not run on hardware
//
#include <hip/hip_runtime.h>
#include <math.h>
#include <stdint.h>

typedef __attribute__((ext_vector_type(16))) _Float16     v16h;
typedef __attribute__((ext_vector_type(8)))  float        v8f;
typedef __attribute__((ext_vector_type(4)))  unsigned int v4u;
typedef __attribute__((ext_vector_type(8)))  int          v8i;
typedef __attribute__((ext_vector_type(4)))  int          v4i;

#define NB     8       // batch
#define CIN    64      // input channels
#define HWIN   128     // input h=w
#define HOUT   64      // output h=w (stride 2)
#define OC     72      // G*K*K dynamic-filter channels
#define KTOT   576     // CIN*K*K GEMM reduction dim
#define MT     5       // ceil(OC/16) WMMA M tiles (padded to 80)
#define KCN    18      // KTOT/32 WMMA K chunks
#define BN_EPS 1e-5f

#define CBLK        (3 * HWIN * 4)              // bytes per channel row-block (1536)
#define APN         (MT * KCN * 32 * 16)        // packed-A f16 elements (46080)
#define XTILE_BYTES (CIN * 3 * HWIN * 4)        // 98304 B staged x tile
#define SIG_BYTES   (4 * OC * 16 * 4)           // 18432 B softmax results
#define SMEM_BYTES  (XTILE_BYTES + SIG_BYTES)   // 116736 B dynamic LDS

// ---------------------------------------------------------------------------
// Prep: pack conv_w (72x576 f32) into per-lane WMMA A fragments (f16) with oc
// rows 72..79 zeroed, and fold BN into scale/shift tables (padded rows get
// scale=0, shift=-inf so they drop out of the softmax with no branches).
// A-fragment halfword map (contiguous K, M = lane%16):
//   lanes  0-15: K=j      (K 0..15)
//   lanes 16-31: K=j+16   (K 16..31)
// ---------------------------------------------------------------------------
__global__ __launch_bounds__(256)
void pasa_prep(const float* __restrict__ w,
               const float* __restrict__ gamma, const float* __restrict__ beta,
               const float* __restrict__ rmean, const float* __restrict__ rvar,
               _Float16* __restrict__ apack,
               float* __restrict__ scale, float* __restrict__ shift) {
    int idx = blockIdx.x * blockDim.x + threadIdx.x;
    if (idx < APN) {
        int j    = idx & 15;
        int lane = (idx >> 4) & 31;
        int kc   = (idx >> 9) % KCN;
        int mt   = (idx >> 9) / KCN;
        int lo = lane & 15;
        int hi = lane >> 4;
        int kk = hi * 16 + j;
        int oc = mt * 16 + lo;
        int k  = kc * 32 + kk;
        float v = (oc < OC) ? w[oc * KTOT + k] : 0.0f;
        apack[idx] = (_Float16)v;
    } else if (idx < APN + MT * 16) {
        int oc = idx - APN;
        if (oc < OC) {
            float s = gamma[oc] * rsqrtf(rvar[oc] + BN_EPS);
            scale[oc] = s;
            shift[oc] = beta[oc] - rmean[oc] * s;
        } else {
            scale[oc] = 0.0f;
            shift[oc] = -INFINITY;
        }
    }
}

// ---------------------------------------------------------------------------
// Fused kernel: one block per (n, ho) output row.  Wave 0 TDM-stages the
// 64ch x 3row x 128col x-tile into LDS; 4 waves each own 16 output positions
// and run the 80x16x576 sigma GEMM with v_wmma_f32_16x16x32_f16, softmax(72)
// via in-register reduction + shfl_xor(16), then the 9-tap f32 weighted sum.
// ---------------------------------------------------------------------------
__global__ __launch_bounds__(128)
void pasa_fused(const float* __restrict__ x,
                const _Float16* __restrict__ apack,
                const float* __restrict__ scale,
                const float* __restrict__ shift,
                float* __restrict__ out) {
    extern __shared__ char smem[];
    float* xtile = (float*)smem;                   // [CIN][3][HWIN] f32
    float* sig   = (float*)(smem + XTILE_BYTES);   // [4][OC][16]   f32

    const int tid  = threadIdx.x;
    const int wave = tid >> 5;
    const int lane = tid & 31;
    const int lo   = lane & 15;
    const int hi   = lane >> 4;

    const int n  = blockIdx.x >> 6;
    const int ho = blockIdx.x & 63;
    const int base_row = (ho == 0) ? 0 : (2 * ho - 1);
    const int wbase = wave * 16;

    // per-tap LDS row index (relative to base_row) and column (reflect: -1 -> 1)
    int ridx[3], rw[3];
#pragma unroll
    for (int kh = 0; kh < 3; ++kh) {
        int i = 2 * ho + kh - 1;
        if (i < 0) i = 1;
        ridx[kh] = i - base_row;      // ho>0: {0,1,2}; ho==0: {1,0,1}
    }
#pragma unroll
    for (int kw = 0; kw < 3; ++kw) {
        int i = 2 * (wbase + lo) + kw - 1;
        rw[kw] = (i < 0) ? 1 : i;
    }
    // 9 tap byte offsets within one channel's 3x128 row block of xtile
    int tap[9];
#pragma unroll
    for (int t = 0; t < 9; ++t) tap[t] = (ridx[t / 3] * HWIN + rw[t % 3]) * 4;

    // Combined per-lane tap table absorbing the hi-half K offset (+16):
    // 16 = 9 + 7, so hi lanes use tap[(t+7)%9] and a channel-block carry of
    // (t >= 2 ? 2 : 1).  After this, every B element is a single ds_load with
    // a compile-time immediate offset d0*1536.
    int ctap[9];
#pragma unroll
    for (int t = 0; t < 9; ++t) {
        int thi = (t + 7) % 9;
        int chi = tap[thi] + ((t >= 2) ? 2 : 1) * CBLK;
        ctap[t] = hi ? chi : tap[t];
    }

    // ---- TDM: DMA x[n, 0:64, base_row:base_row+3, 0:128] into LDS ----------
    if (wave == 0) {
        uint64_t ga = (uint64_t)(uintptr_t)(x + ((size_t)n * CIN * HWIN + (size_t)base_row) * HWIN);
        unsigned ldsb = __builtin_amdgcn_groupstaticsize();  // dynamic LDS base; xtile at +0
        v4u g0;
        g0[0] = 1u;                                              // count=1, load, user mode
        g0[1] = ldsb;                                            // lds_addr (bytes)
        g0[2] = (unsigned)ga;                                    // global_addr[31:0]
        g0[3] = (unsigned)((ga >> 32) & 0x1FFFFFFu) | (2u << 30);// global_addr[56:32], type=2
        v8i g1;
        g1[0] = (int)(2u << 16);                 // data_size = 2 (4 bytes)
        g1[1] = (int)((unsigned)HWIN << 16);     // tensor_dim0 = 128 (low 16)
        g1[2] = (int)(3u << 16);                 // dim0 hi16 = 0 | tensor_dim1 = 3
        g1[3] = (int)((unsigned)HWIN << 16);     // dim1 hi16 = 0 | tile_dim0 = 128
        g1[4] = (int)(3u | (64u << 16));         // tile_dim1 = 3 | tile_dim2 = 64
        g1[5] = HWIN;                            // tensor_dim0_stride = 128 (row stride)
        g1[6] = (int)(((unsigned)(HWIN * HWIN) & 0xFFFFu) << 16); // dim1_stride low16 (16384)
        g1[7] = (int)((unsigned)(HWIN * HWIN) >> 16);             // dim1_stride [47:16]
        v4i g2;
        g2[0] = CIN;                             // tensor_dim2 = 64
        g2[1] = 0; g2[2] = 0; g2[3] = 0;         // dim3 / dim2_stride / tile_dim3 unused
        v4i g3;
        g3[0] = 0; g3[1] = 0; g3[2] = 0; g3[3] = 0;
        v8i gx;                                  // extended/reserved operand (6-arg form)
        gx[0] = 0; gx[1] = 0; gx[2] = 0; gx[3] = 0;
        gx[4] = 0; gx[5] = 0; gx[6] = 0; gx[7] = 0;
        __builtin_amdgcn_tensor_load_to_lds(g0, g1, g2, g3, gx, 0);
        __builtin_amdgcn_s_wait_tensorcnt(0);
    }
    __syncthreads();

    const char* xb = (const char*)xtile;

    // ---- sigma GEMM: fully unrolled K loop, all tap indices compile-time ----
    v8f acc[MT];
#pragma unroll
    for (int mt = 0; mt < MT; ++mt) acc[mt] = (v8f){};

#pragma unroll
    for (int kc = 0; kc < KCN; ++kc) {
        v16h b;
#pragma unroll
        for (int j = 0; j < 16; ++j) {
            const int k0 = kc * 32 + j;        // K index for hi==0 lanes
            const int t0 = k0 % 9;             // tap selector (compile-time)
            const int d0 = (k0 / 9) * CBLK;    // channel-block immediate
            b[j] = (_Float16)(*(const float*)(xb + d0 + ctap[t0]));
        }
#pragma unroll
        for (int mt = 0; mt < MT; ++mt) {
            v16h a = ((const v16h*)apack)[(mt * KCN + kc) * 32 + lane];
            acc[mt] = __builtin_amdgcn_wmma_f32_16x16x32_f16(
                false, a, false, b, (short)0, acc[mt], false, false);
        }
    }

    // ---- BN + softmax over 72 channels of column p = lo ---------------------
    // C layout: acc[mt] VGPR r = (M = r + hi*8, N = lo) -> oc = mt*16 + hi*8 + r
    float vals[MT * 8];
    float m = -INFINITY;
#pragma unroll
    for (int mt = 0; mt < MT; ++mt) {
#pragma unroll
        for (int r = 0; r < 8; ++r) {
            int oc = mt * 16 + hi * 8 + r;
            float v = acc[mt][r] * scale[oc] + shift[oc];  // padded rows -> -inf
            vals[mt * 8 + r] = v;
            m = fmaxf(m, v);
        }
    }
    m = fmaxf(m, __shfl_xor(m, 16, 32));
    float ssum = 0.0f;
#pragma unroll
    for (int i = 0; i < MT * 8; ++i) {
        float e = __expf(vals[i] - m);
        vals[i] = e;
        ssum += e;
    }
    ssum += __shfl_xor(ssum, 16, 32);
    float sinv = 1.0f / ssum;

#pragma unroll
    for (int mt = 0; mt < MT; ++mt) {
#pragma unroll
        for (int r = 0; r < 8; ++r) {
            int oc = mt * 16 + hi * 8 + r;
            if (oc < OC) sig[(wave * OC + oc) * 16 + lo] = vals[mt * 8 + r] * sinv;
        }
    }
    __syncthreads();

    // ---- 9-tap group weighted sum (f32, from LDS); lane owns p = lo --------
    const int wo = wbase + lo;
    int otap[9];                                    // taps offset to channel half
#pragma unroll
    for (int t = 0; t < 9; ++t) otap[t] = hi * (32 * CBLK) + tap[t];
    const float* sigp = sig + (wave * OC + hi * 36) * 16 + lo;  // g = hi*4 + cc/8
#pragma unroll
    for (int cc = 0; cc < 32; ++cc) {
        float sum = 0.0f;
#pragma unroll
        for (int k = 0; k < 9; ++k) {
            float s  = sigp[((cc >> 3) * 9 + k) * 16];               // imm <= 2240
            float xv = *(const float*)(xb + cc * CBLK + otap[k]);    // imm <= 47616
            sum = fmaf(s, xv, sum);
        }
        out[(((size_t)n * CIN + hi * 32 + cc) * HOUT + ho) * HOUT + wo] = sum;
    }
}

extern "C" void kernel_launch(void* const* d_in, const int* in_sizes, int n_in,
                              void* d_out, int out_size, void* d_ws, size_t ws_size,
                              hipStream_t stream) {
    const float* x     = (const float*)d_in[0];
    const float* w     = (const float*)d_in[1];
    const float* gamma = (const float*)d_in[2];
    const float* beta  = (const float*)d_in[3];
    const float* rmean = (const float*)d_in[4];
    const float* rvar  = (const float*)d_in[5];
    float* out = (float*)d_out;

    _Float16* apack = (_Float16*)d_ws;                       // 92,160 B
    float* scale = (float*)((char*)d_ws + APN * 2);          //    320 B
    float* shift = scale + MT * 16;                          //    320 B

    const int prep_total = APN + MT * 16;
    pasa_prep<<<(prep_total + 255) / 256, 256, 0, stream>>>(
        w, gamma, beta, rmean, rvar, apack, scale, shift);

    // one block per (n, ho): 8 * 64 = 512 blocks, 4 waves each
    pasa_fused<<<NB * HOUT, 128, SMEM_BYTES, stream>>>(x, apack, scale, shift, out);
}
